// Qwen3MoeSparseFeedForward_4002909519902
// MI455X (gfx1250) — matrix operand order repaired and checked by reading, not compiled, not run on hardware
//
#include <hip/hip_runtime.h>
#include <hip/hip_bf16.h>
#include <stdint.h>

// ---------------------------------------------------------------------------
// Qwen3 MoE sparse FFN for MI455X (gfx1250, wave32, WMMA).
// T=2048 tokens, H=1024, I=512, E=16 experts, top-4 routing, gated SiLU.
// router -> per-expert token buckets -> per-(expert, 32-token tile) fused FFN
// using v_wmma_f32_16x16x32_bf16; fp32->bf16 via single v_perm_b32 per pair.
// B fragments software-pipelined (prefetch t+1 while t's WMMAs issue) so the
// two buffers are simultaneously live -> distinct regs -> no WAR v_nops.
// ---------------------------------------------------------------------------

typedef __attribute__((ext_vector_type(16))) __bf16 v16bf;
typedef __attribute__((ext_vector_type(8)))  float  v8f;

#define T_TOK 2048
#define H_DIM 1024
#define I_DIM 512
#define E_NUM 16
#define TOPK  4
#define CAP   2048   // max tokens routed to one expert (worst case = T)
#define MT    32     // token rows per FFN tile (2 WMMA row-tiles share each B frag)

union AV { v16bf v; unsigned u[8]; };

// pack bf16(f0) | bf16(f1)<<16 in ONE v_perm_b32 (truncating convert)
__device__ __forceinline__ unsigned packbf(float f0, float f1) {
  return __builtin_amdgcn_perm(__builtin_bit_cast(unsigned, f1),
                               __builtin_bit_cast(unsigned, f0), 0x07060302u);
}

// B fragment: 16 consecutive f32 K-values for this lane -> 16 bf16
__device__ __forceinline__ void load_b_frag(const float* __restrict__ p, AV& b) {
  const float4* p4 = (const float4*)p;
#pragma unroll
  for (int q = 0; q < 4; ++q) {
    float4 f = p4[q];
    b.u[2 * q]     = packbf(f.x, f.y);
    b.u[2 * q + 1] = packbf(f.z, f.w);
  }
}

// A fragment from LDS bf16 row: contiguous run [ahalf..ahalf+15]
__device__ __forceinline__ void load_a_frag(const unsigned short* lp, int ahalf, AV& a) {
  uint4 lo = *(const uint4*)(lp + ahalf);
  uint4 hi = *(const uint4*)(lp + ahalf + 8);
  a.u[0] = lo.x; a.u[1] = lo.y; a.u[2] = lo.z; a.u[3] = lo.w;
  a.u[4] = hi.x; a.u[5] = hi.y; a.u[6] = hi.z; a.u[7] = hi.w;
}

#define WMMA_BF16(A, B, C) \
  __builtin_amdgcn_wmma_f32_16x16x32_bf16(false, (A).v, false, (B).v, (short)0, (C), false, false)

// ---------------- kernel 0: zero the (poisoned) output ----------------
__global__ void moe_zero_out(float* __restrict__ out) {
  int idx = blockIdx.x * blockDim.x + threadIdx.x;
  float4 z; z.x = 0.f; z.y = 0.f; z.z = 0.f; z.w = 0.f;
  ((float4*)out)[idx] = z;
}

// ---------------- kernel 1: router (one wave32 per token) ----------------
__global__ void moe_router(const float* __restrict__ x, const float* __restrict__ gw,
                           int* __restrict__ topk_id, float* __restrict__ topk_w) {
  const int lane = threadIdx.x & 31;
  const int wv   = threadIdx.x >> 5;
  const int t    = blockIdx.x * 4 + wv;

  float acc[E_NUM];
#pragma unroll
  for (int e = 0; e < E_NUM; ++e) acc[e] = 0.f;

  const float* xr = x + (size_t)t * H_DIM;
  for (int h = lane; h < H_DIM; h += 32) {
    float xv = xr[h];
#pragma unroll
    for (int e = 0; e < E_NUM; ++e) acc[e] += xv * gw[e * H_DIM + h];
  }
#pragma unroll
  for (int off = 16; off >= 1; off >>= 1) {
#pragma unroll
    for (int e = 0; e < E_NUM; ++e) acc[e] += __shfl_xor(acc[e], off, 32);
  }

  if (lane == 0) {
    float mx = acc[0];
#pragma unroll
    for (int e = 1; e < E_NUM; ++e) mx = fmaxf(mx, acc[e]);
    unsigned used = 0;
    int ids[TOPK]; float vs[TOPK]; float s = 0.f;
#pragma unroll
    for (int k = 0; k < TOPK; ++k) {
      float best = -3.4e38f; int bi = 0;
#pragma unroll
      for (int e = 0; e < E_NUM; ++e)
        if (!(used & (1u << e)) && acc[e] > best) { best = acc[e]; bi = e; }
      used |= (1u << bi);
      ids[k] = bi;
      vs[k]  = __expf(best - mx);   // renormed top-k softmax: denom cancels
      s += vs[k];
    }
    float inv = 1.f / s;
#pragma unroll
    for (int k = 0; k < TOPK; ++k) {
      topk_id[t * TOPK + k] = ids[k];
      topk_w [t * TOPK + k] = vs[k] * inv;
    }
  }
}

// ---------------- kernel 2: bucket tokens per expert (1 block) ----------------
__global__ void moe_scatter(const int* __restrict__ topk_id, const float* __restrict__ topk_w,
                            int* __restrict__ counts, int* __restrict__ btok,
                            float* __restrict__ bw) {
  __shared__ int cnt[E_NUM];
  __shared__ int cur[E_NUM];
  const int tid = threadIdx.x;
  if (tid < E_NUM) { cnt[tid] = 0; cur[tid] = 0; }
  __syncthreads();
  for (int i = tid; i < T_TOK * TOPK; i += 256) atomicAdd(&cnt[topk_id[i]], 1);
  __syncthreads();
  if (tid < E_NUM) counts[tid] = cnt[tid];
  for (int i = tid; i < T_TOK * TOPK; i += 256) {
    int e = topk_id[i];
    int pos = atomicAdd(&cur[e], 1);
    btok[e * CAP + pos] = i >> 2;       // token index
    bw  [e * CAP + pos] = topk_w[i];
  }
}

// ---------------- kernel 3: fused gated-SiLU FFN per (expert, 32-token tile) ----------------
__launch_bounds__(256)
__global__ void moe_ffn(const float* __restrict__ x,
                        const float* __restrict__ w1, const float* __restrict__ w3,
                        const float* __restrict__ w2,
                        const int* __restrict__ counts,
                        const int* __restrict__ btok, const float* __restrict__ bw,
                        float* __restrict__ out) {
  const int e    = blockIdx.y;
  const int cnt  = counts[e];
  const int base = blockIdx.x * MT;
  if (base >= cnt) return;

  __shared__ unsigned short lds_x[MT * H_DIM];    // 64 KB bf16 X tile  [32, H]
  __shared__ unsigned short lds_act[MT * I_DIM];  // 32 KB bf16 act tile [32, I]
  __shared__ int   lds_tok[MT];
  __shared__ float lds_wgt[MT];

  const int tid = threadIdx.x;
  if (tid < MT) {
    int idx = base + tid;
    int tok = 0; float wgt = 0.f;
    if (idx < cnt) { tok = btok[e * CAP + idx]; wgt = bw[e * CAP + idx]; }
    lds_tok[tid] = tok; lds_wgt[tid] = wgt;    // padded rows: token 0, weight 0
  }
  __syncthreads();
  // fill X tile: float4 global load -> 2 perms -> 8B LDS store
  for (int i = tid; i < MT * H_DIM / 4; i += 256) {
    int elem = i << 2;
    int m = elem >> 10, h = elem & (H_DIM - 1);
    float4 f = *(const float4*)(x + (size_t)lds_tok[m] * H_DIM + h);
    uint2 p; p.x = packbf(f.x, f.y); p.y = packbf(f.z, f.w);
    *(uint2*)&lds_x[elem] = p;
  }
  __syncthreads();

  const int lane  = tid & 31, wv = tid >> 5;   // 8 waves
  const int mrow  = lane & 15;                 // A row / B col / C col
  const int ahalf = (lane & 16) ? 16 : 0;      // A fragment K offset (contiguous, matches B)
  const int bbase = (lane & 16) ? 16 : 0;      // B fragment K offset (contiguous k = 16h + s, matches A)
  const int mput  = (lane >> 4) << 3;          // C/D: M = vgpr + (lane>=16 ? 8 : 0)

  // ---- GEMM1: h1/h3[32, I] = Xbf16[32, H] x {W1,W3}[e]^T ----
  v8f acc1[2][4], acc3[2][4];
#pragma unroll
  for (int mt = 0; mt < 2; ++mt)
#pragma unroll
    for (int t = 0; t < 4; ++t)
#pragma unroll
      for (int j = 0; j < 8; ++j) { acc1[mt][t][j] = 0.f; acc3[mt][t][j] = 0.f; }

  const float* w1p = w1 + (size_t)e * I_DIM * H_DIM;
  const float* w3p = w3 + (size_t)e * I_DIM * H_DIM;
  const int n0 = wv * 64;   // 8 waves x 64 cols = I_DIM
  const float* pB1[4]; const float* pB3[4];
#pragma unroll
  for (int t = 0; t < 4; ++t) {
    int n = n0 + t * 16 + mrow;
    pB1[t] = w1p + (size_t)n * H_DIM + bbase;
    pB3[t] = w3p + (size_t)n * H_DIM + bbase;
  }

  {
    AV b1buf[2], b3buf[2];
    // prologue: fragment (k0=0, t=0) into buffer 0
    load_b_frag(pB1[0], b1buf[0]);
    load_b_frag(pB3[0], b3buf[0]);
    for (int k0 = 0; k0 < H_DIM; k0 += 32) {
      AV a0, a1;
      load_a_frag(&lds_x[(mrow << 10) + k0], ahalf, a0);
      load_a_frag(&lds_x[((mrow + 16) << 10) + k0], ahalf, a1);
      const int knext = (k0 + 32) & (H_DIM - 1);   // wrapped tail prefetch (harmless)
#pragma unroll
      for (int t = 0; t < 4; ++t) {
        const int cur = t & 1, nxt = cur ^ 1;
        // prefetch fragment t+1 (or next k-step's t=0) into the other buffer
        const int tn = (t < 3) ? t + 1 : 0;
        const int kn = (t < 3) ? k0 : knext;
        load_b_frag(pB1[tn] + kn, b1buf[nxt]);
        load_b_frag(pB3[tn] + kn, b3buf[nxt]);
        acc1[0][t] = WMMA_BF16(a0, b1buf[cur], acc1[0][t]);
        acc1[1][t] = WMMA_BF16(a1, b1buf[cur], acc1[1][t]);
        acc3[0][t] = WMMA_BF16(a0, b3buf[cur], acc3[0][t]);
        acc3[1][t] = WMMA_BF16(a1, b3buf[cur], acc3[1][t]);
      }
    }
  }

  // ---- epilogue 1: act = silu(h1) * h3 -> LDS (bf16 trunc) ----
#pragma unroll
  for (int mt = 0; mt < 2; ++mt) {
#pragma unroll
    for (int t = 0; t < 4; ++t) {
      int n = n0 + t * 16 + mrow;
#pragma unroll
      for (int v = 0; v < 8; ++v) {
        float h1 = acc1[mt][t][v], h3 = acc3[mt][t][v];
        float act = (h1 / (1.f + __expf(-h1))) * h3;
        int m = mt * 16 + mput + v;
        lds_act[(m << 9) + n] =
            (unsigned short)(__builtin_bit_cast(unsigned, act) >> 16);
      }
    }
  }
  __syncthreads();

  // ---- GEMM2: out[32, H] = act[32, I] x W2[e]^T ----
  v8f acco[2][8];
#pragma unroll
  for (int mt = 0; mt < 2; ++mt)
#pragma unroll
    for (int t = 0; t < 8; ++t)
#pragma unroll
      for (int j = 0; j < 8; ++j) acco[mt][t][j] = 0.f;

  const float* w2p = w2 + (size_t)e * H_DIM * I_DIM;
  const int h0w = wv * 128;   // 8 waves x 128 cols = H_DIM
  const float* pB2[8];
#pragma unroll
  for (int t = 0; t < 8; ++t) {
    int hcol = h0w + t * 16 + mrow;
    pB2[t] = w2p + (size_t)hcol * I_DIM + bbase;
  }

  {
    AV bbuf[2];
    load_b_frag(pB2[0], bbuf[0]);   // prologue
    for (int k0 = 0; k0 < I_DIM; k0 += 32) {
      AV a0, a1;
      load_a_frag(&lds_act[(mrow << 9) + k0], ahalf, a0);
      load_a_frag(&lds_act[((mrow + 16) << 9) + k0], ahalf, a1);
      const int knext = (k0 + 32) & (I_DIM - 1);
#pragma unroll
      for (int t = 0; t < 8; ++t) {
        const int cur = t & 1, nxt = cur ^ 1;
        const int tn = (t < 7) ? t + 1 : 0;
        const int kn = (t < 7) ? k0 : knext;
        load_b_frag(pB2[tn] + kn, bbuf[nxt]);
        acco[0][t] = WMMA_BF16(a0, bbuf[cur], acco[0][t]);
        acco[1][t] = WMMA_BF16(a1, bbuf[cur], acco[1][t]);
      }
    }
  }

  // ---- epilogue 2: scale by routing weight, accumulate into out ----
#pragma unroll
  for (int mt = 0; mt < 2; ++mt) {
#pragma unroll
    for (int v = 0; v < 8; ++v) {
      int m = mt * 16 + mput + v;
      float wgt = lds_wgt[m];
      float* rowp = out + (size_t)lds_tok[m] * H_DIM + mrow + h0w;
#pragma unroll
      for (int t = 0; t < 8; ++t) {
        __hip_atomic_fetch_add(rowp + t * 16, acco[mt][t][v] * wgt,
                               __ATOMIC_RELAXED, __HIP_MEMORY_SCOPE_AGENT);
      }
    }
  }
}

// ---------------- launch ----------------
extern "C" void kernel_launch(void* const* d_in, const int* in_sizes, int n_in,
                              void* d_out, int out_size, void* d_ws, size_t ws_size,
                              hipStream_t stream) {
  (void)in_sizes; (void)n_in; (void)out_size; (void)ws_size;
  const float* x  = (const float*)d_in[0];
  const float* gw = (const float*)d_in[1];
  const float* w1 = (const float*)d_in[2];
  const float* w3 = (const float*)d_in[3];
  const float* w2 = (const float*)d_in[4];
  float* out = (float*)d_out;

  char* ws = (char*)d_ws;
  int*   topk_id = (int*)  (ws + 0);                          // 32 KB
  float* topk_w  = (float*)(ws + 32768);                      // 32 KB
  int*   counts  = (int*)  (ws + 65536);                      // 64 B
  int*   btok    = (int*)  (ws + 66560);                      // 128 KB
  float* bw      = (float*)(ws + 66560 + E_NUM * CAP * 4);    // 128 KB

  moe_zero_out<<<(T_TOK * H_DIM / 4) / 256, 256, 0, stream>>>(out);
  moe_router  <<<T_TOK / 4, 128, 0, stream>>>(x, gw, topk_id, topk_w);
  moe_scatter <<<1, 256, 0, stream>>>(topk_id, topk_w, counts, btok, bw);
  moe_ffn     <<<dim3(CAP / MT, E_NUM), 256, 0, stream>>>(x, w1, w3, w2, counts, btok, bw, out);
}
